// GraphAdapter_56916906606893
// MI455X (gfx1250) — matrix operand order repaired and checked by reading, not compiled, not run on hardware
//
#include <hip/hip_runtime.h>
#include <hip/hip_bf16.h>

// ---------------------------------------------------------------------------
// MI455X (gfx1250): bf16 WMMA 16x16x32 (f32 accum) for all GEMMs.
//  - MLP g(z) factored to node level (20000 rows) instead of per pair (320000).
//  - Weights pre-swizzled into WMMA B-fragment order: B fragments are single
//    32B contiguous per-lane global loads (L2-resident, 192MB L2).
//  - Full-K A tile staged in LDS ONCE per block (66KB of the 320KB/WGP pool):
//    one barrier total, barrier-free WMMA main loop.
//  - Edge kernel: per-row gather base pointers precomputed into LDS; staging
//    is pure b64/b128 loads + packed mul-add + bf16 pack.
// ---------------------------------------------------------------------------

typedef __bf16 bf16;
typedef __attribute__((ext_vector_type(16))) __bf16 v16bf;
typedef __attribute__((ext_vector_type(8)))  __bf16 v8bf;
typedef __attribute__((ext_vector_type(4)))  __bf16 v4bf;
typedef __attribute__((ext_vector_type(8)))  float  v8f;

#define N_NODES 20000
#define E_EDGES 320000
#define DIM     256
#define VOCAB   512
#define KTILES  (DIM / 32)

#define LDK 264  // bf16 row stride for 128x256 A tiles (528B = 33*16: aligned,
                 // 132 words % 64 banks = 4 -> 16 lanes hit 16 distinct banks)

static __device__ inline v8f wmma_bf16(v16bf a, v16bf b, v8f c) {
    return __builtin_amdgcn_wmma_f32_16x16x32_bf16(
        false, a, false, b, (short)0, c, false, false);
}

// A fragment (16x32 bf16) at k-offset kt from the full-K LDS tile.
// Per-lane data: two contiguous 16-byte runs -> 2x ds_load_b128 + shuffle.
// lanes 0-15: M=lane, elems 0..7=K0..7, 8..15=K16..23;
// lanes 16-31: M=lane-16, elems 0..7=K8..15, 8..15=K24..31.
static __device__ inline v16bf load_a_frag(const bf16* s, int kt, int m0) {
    int lane = threadIdx.x & 31;
    const bf16* row = s + (size_t)(m0 + (lane & 15)) * LDK + kt + ((lane < 16) ? 0 : 8);
    v8bf lo = *(const v8bf*)(row);
    v8bf hi = *(const v8bf*)(row + 16);
    return __builtin_shufflevector(lo, hi, 0, 1, 2, 3, 4, 5, 6, 7,
                                           8, 9, 10, 11, 12, 13, 14, 15);
}

// B fragment (32x16 bf16) direct from pre-swizzled global weights:
// one 32-byte contiguous load per lane.
static __device__ inline v16bf load_b_frag_g(const bf16* __restrict__ Wsw,
                                             int kt, int n0) {
    int lane = threadIdx.x & 31;
    const bf16* p = Wsw + ((size_t)(n0 >> 4) * KTILES + (kt >> 5)) * 512 + lane * 16;
    return *(const v16bf*)p;
}

// ---------------------------------------------------------------------------
// fp32 -> bf16 (row-major, for z)
// ---------------------------------------------------------------------------
__global__ __launch_bounds__(256) void f32_to_bf16_kernel(const float* __restrict__ in,
                                                          bf16* __restrict__ out, int n) {
    int i = blockIdx.x * 256 + threadIdx.x;
    if (i < n) out[i] = (bf16)in[i];
}

// ---------------------------------------------------------------------------
// Swizzle fp32 weight [K,N] into bf16 WMMA B-fragment order:
// tile (t=k/32, j=n/16) at ((j*KTILES)+t)*512; within tile:
// lane = (n%16) + 16*((k/8)%2), elem = k%8 + 8*(k%32>=16).
// ---------------------------------------------------------------------------
__global__ __launch_bounds__(256) void swizzle_weight_kernel(const float* __restrict__ W,
                                                             bf16* __restrict__ out,
                                                             int K, int N) {
    int i = blockIdx.x * 256 + threadIdx.x;
    if (i >= K * N) return;
    int k = i / N, n = i % N;
    int t = k >> 5, j = n >> 4;
    int kl = k & 31, nl = n & 15;
    int lane = nl + ((kl & 8) << 1);
    int e = (kl & 7) + ((kl & 16) >> 1);
    out[((size_t)j * (K >> 5) + t) * 512 + lane * 16 + e] = (bf16)W[i];
}

// ---------------------------------------------------------------------------
// Node GEMM: C[M,256] = act(A_bf16[M,256] @ Bsw + bias).
// Block tile 128x64, 8 waves (one 16-row slab each, 4 N-tiles).
// Full-K A staged once; barrier-free main loop; B straight from L2.
// ---------------------------------------------------------------------------
template<bool RELU, bool OUT_BF16>
__global__ __launch_bounds__(256) void gemm_node_kernel(
    const bf16* __restrict__ A, const bf16* __restrict__ Bsw,
    const float* __restrict__ bias, void* __restrict__ Cptr, int M)
{
    __shared__ alignas(16) bf16 As[128 * LDK];   // 66 KB

    const int tid     = threadIdx.x;
    const int wave    = tid >> 5;
    const int lane    = tid & 31;
    const int rowBase = blockIdx.x * 128;
    const int colBase = blockIdx.y * 64;

    // Stage full 128x256 A tile: 16B vector copies, one wave per row,
    // 512B contiguous global reads per wave.
    for (int i = tid; i < 4096; i += 256) {
        int r = i >> 5;              // 32 chunks of 8 bf16 per row
        int c = (i & 31) * 8;
        int gr = rowBase + r;
        v8bf val = {};
        if (gr < M) val = *(const v8bf*)(A + (size_t)gr * DIM + c);
        *(v8bf*)(As + r * LDK + c) = val;
    }
    __syncthreads();

    v8f acc[4];
#pragma unroll
    for (int j = 0; j < 4; ++j) acc[j] = (v8f)0.0f;

    for (int kt = 0; kt < DIM; kt += 32) {
        v16bf af = load_a_frag(As, kt, wave * 16);
#pragma unroll
        for (int j = 0; j < 4; ++j) {
            v16bf bfrag = load_b_frag_g(Bsw, kt, colBase + j * 16);
            acc[j] = wmma_bf16(af, bfrag, acc[j]);
        }
    }

    // Epilogue. C/D layout: row = v + 8*(lane>=16), col = lane&15.
    const int rbase = rowBase + wave * 16 + ((lane < 16) ? 0 : 8);
#pragma unroll
    for (int j = 0; j < 4; ++j) {
        int col  = colBase + j * 16 + (lane & 15);
        float bb = bias[col];
#pragma unroll
        for (int v = 0; v < 8; ++v) {
            int r = rbase + v;
            if (r < M) {
                float val = acc[j][v] + bb;
                if (RELU) val = fmaxf(val, 0.0f);
                if (OUT_BF16) ((bf16*)Cptr)[(size_t)r * DIM + col] = (bf16)val;
                else          ((float*)Cptr)[(size_t)r * DIM + col] = val;
            }
        }
    }
}

// ---------------------------------------------------------------------------
// Attention gate: a[p] = sigmoid(dot(q[dst_p], k[src_p])). One wave per pair,
// float4 loads (8 floats per lane), shuffle reduction.
// ---------------------------------------------------------------------------
__global__ __launch_bounds__(256) void attn_kernel(
    const float* __restrict__ q, const float* __restrict__ k,
    const int* __restrict__ src, const int* __restrict__ dst,
    float* __restrict__ a, int E)
{
    int wid  = (blockIdx.x * 256 + threadIdx.x) >> 5;
    int lane = threadIdx.x & 31;
    if (wid >= E) return;
    const float4* qr = (const float4*)(q + (size_t)dst[wid] * DIM);
    const float4* kr = (const float4*)(k + (size_t)src[wid] * DIM);
    float4 q0 = qr[lane * 2], q1 = qr[lane * 2 + 1];
    float4 k0 = kr[lane * 2], k1 = kr[lane * 2 + 1];
    float sum = q0.x * k0.x + q0.y * k0.y + q0.z * k0.z + q0.w * k0.w +
                q1.x * k1.x + q1.y * k1.y + q1.z * k1.z + q1.w * k1.w;
#pragma unroll
    for (int off = 16; off > 0; off >>= 1) sum += __shfl_xor(sum, off, 32);
    if (lane == 0) a[wid] = 1.0f / (1.0f + expf(-sum));
}

// ---------------------------------------------------------------------------
// Fused edge kernel: h_tilde rows built once into a full-K LDS tile
// (precomputed per-row base pointers; float4 gathers; packed bf16 stores),
// then a barrier-free WMMA loop against pre-swizzled Wpred.
// Block 128 rows x 128 cols, 8 waves (4M x 2N), 2x4 acc tiles per wave.
// E=320000, V=512: exact tiling, no guards.
// ---------------------------------------------------------------------------
__global__ __launch_bounds__(256) void edge_logits_kernel(
    const float* __restrict__ h_mask, const float* __restrict__ gz,
    const float* __restrict__ a,
    const int* __restrict__ src, const int* __restrict__ dst,
    const bf16* __restrict__ WpredSw, const float* __restrict__ bpred,
    float* __restrict__ out)
{
    __shared__ float              sA[128];
    __shared__ unsigned long long sHP[128];   // &h_mask[dst[p]*DIM]
    __shared__ unsigned long long sGP[128];   // &gz[src[p]*DIM]
    __shared__ alignas(16) bf16   As[128 * LDK];   // 66 KB

    const int tid     = threadIdx.x;
    const int rowBase = blockIdx.x * 128;   // pair index base
    const int colBase = blockIdx.y * 128;   // vocab column base

    if (tid < 128) {
        int p = rowBase + tid;
        sA[tid]  = a[p];
        sHP[tid] = (unsigned long long)(h_mask + (size_t)dst[p] * DIM);
        sGP[tid] = (unsigned long long)(gz    + (size_t)src[p] * DIM);
    }
    __syncthreads();

    // Stage full 128x256 h_tilde tile once: 64 consecutive lanes walk one row
    // (1KB contiguous global reads per row pair of waves).
    for (int i = tid; i < 8192; i += 256) {
        int r = i >> 6;              // 64 chunks of 4 floats per row
        int c = (i & 63) * 4;
        float av = sA[r];
        const float* hp = (const float*)sHP[r];
        const float* gp = (const float*)sGP[r];
        float4 h4 = *(const float4*)(hp + c);
        float4 g4 = *(const float4*)(gp + c);
        v4bf pk;
        pk[0] = (bf16)(av * h4.x + (1.0f - av) * g4.x);
        pk[1] = (bf16)(av * h4.y + (1.0f - av) * g4.y);
        pk[2] = (bf16)(av * h4.z + (1.0f - av) * g4.z);
        pk[3] = (bf16)(av * h4.w + (1.0f - av) * g4.w);
        *(v4bf*)(As + r * LDK + c) = pk;
    }
    __syncthreads();

    const int wave = tid >> 5;
    const int lane = tid & 31;
    const int wm = wave & 3;   // 4 waves in M: 32 rows each
    const int wn = wave >> 2;  // 2 waves in N: 64 cols each

    v8f acc[2][4];
#pragma unroll
    for (int mi = 0; mi < 2; ++mi)
#pragma unroll
        for (int j = 0; j < 4; ++j) acc[mi][j] = (v8f)0.0f;

    // Barrier-free main loop: 8 k-steps x 8 WMMAs.
    for (int kt = 0; kt < DIM; kt += 32) {
        v16bf af0 = load_a_frag(As, kt, wm * 32);
        v16bf af1 = load_a_frag(As, kt, wm * 32 + 16);
#pragma unroll
        for (int j = 0; j < 4; ++j) {
            v16bf bfrag = load_b_frag_g(WpredSw, kt, colBase + wn * 64 + j * 16);
            acc[0][j] = wmma_bf16(af0, bfrag, acc[0][j]);
            acc[1][j] = wmma_bf16(af1, bfrag, acc[1][j]);
        }
    }

    // Epilogue: += bpred, store fp32 logits
#pragma unroll
    for (int mi = 0; mi < 2; ++mi) {
        int r0 = rowBase + wm * 32 + mi * 16 + ((lane < 16) ? 0 : 8);
#pragma unroll
        for (int j = 0; j < 4; ++j) {
            int col  = colBase + wn * 64 + j * 16 + (lane & 15);
            float bb = bpred[col];
#pragma unroll
            for (int v = 0; v < 8; ++v)
                out[(size_t)(r0 + v) * VOCAB + col] = acc[mi][j][v] + bb;
        }
    }
}

__global__ __launch_bounds__(256) void copy_int_kernel(const int* __restrict__ in,
                                                       int* __restrict__ out, int n) {
    int i = blockIdx.x * 256 + threadIdx.x;
    if (i < n) out[i] = in[i];
}

// ---------------------------------------------------------------------------
// Launch
// ---------------------------------------------------------------------------
extern "C" void kernel_launch(void* const* d_in, const int* in_sizes, int n_in,
                              void* d_out, int out_size, void* d_ws, size_t ws_size,
                              hipStream_t stream) {
    (void)in_sizes; (void)n_in; (void)out_size; (void)ws_size;

    const float* h_mask = (const float*)d_in[0];
    const float* z      = (const float*)d_in[1];
    const float* Wq     = (const float*)d_in[2];
    const float* bq     = (const float*)d_in[3];
    const float* Wk     = (const float*)d_in[4];
    const float* bk     = (const float*)d_in[5];
    const float* W1     = (const float*)d_in[6];
    const float* b1     = (const float*)d_in[7];
    const float* W2     = (const float*)d_in[8];
    const float* b2     = (const float*)d_in[9];
    const float* Wpred  = (const float*)d_in[10];
    const float* bpred  = (const float*)d_in[11];
    const int*   edge_index = (const int*)d_in[12];  // [2, E]
    const int*   masked_idx = (const int*)d_in[14];  // == edge_index[1]
    const int*   src = edge_index;                    // row 0: neighbor nodes
    const int*   dst = masked_idx;                    // center / masked nodes

    float* out = (float*)d_out;

    // Carve workspace (~85 MB)
    char* ws = (char*)d_ws;
    auto carve = [&](size_t bytes) -> char* {
        char* p = ws;
        ws += (bytes + 255) & ~(size_t)255;
        return p;
    };
    const size_t ND = (size_t)N_NODES * DIM;
    bf16*  zb  = (bf16*) carve(ND * sizeof(bf16));
    bf16*  Wqs = (bf16*) carve((size_t)DIM * DIM * sizeof(bf16));
    bf16*  Wks = (bf16*) carve((size_t)DIM * DIM * sizeof(bf16));
    bf16*  W1s = (bf16*) carve((size_t)DIM * DIM * sizeof(bf16));
    bf16*  W2s = (bf16*) carve((size_t)DIM * DIM * sizeof(bf16));
    bf16*  Wps = (bf16*) carve((size_t)DIM * VOCAB * sizeof(bf16));
    float* qf  = (float*)carve(ND * sizeof(float));
    float* kf  = (float*)carve(ND * sizeof(float));
    bf16*  tb  = (bf16*) carve(ND * sizeof(bf16));
    float* gzf = (float*)carve(ND * sizeof(float));
    float* af  = (float*)carve((size_t)E_EDGES * sizeof(float));

    // 1) z -> bf16 row-major; weights -> bf16 WMMA-B-fragment order
    f32_to_bf16_kernel<<<(int)((ND + 255) / 256), 256, 0, stream>>>(z, zb, (int)ND);
    swizzle_weight_kernel<<<(DIM * DIM + 255) / 256, 256, 0, stream>>>(Wq, Wqs, DIM, DIM);
    swizzle_weight_kernel<<<(DIM * DIM + 255) / 256, 256, 0, stream>>>(Wk, Wks, DIM, DIM);
    swizzle_weight_kernel<<<(DIM * DIM + 255) / 256, 256, 0, stream>>>(W1, W1s, DIM, DIM);
    swizzle_weight_kernel<<<(DIM * DIM + 255) / 256, 256, 0, stream>>>(W2, W2s, DIM, DIM);
    swizzle_weight_kernel<<<(DIM * VOCAB + 255) / 256, 256, 0, stream>>>(Wpred, Wps, DIM, VOCAB);

    // 2) Node-level projections (WMMA bf16)
    dim3 gNode((N_NODES + 127) / 128, DIM / 64);
    gemm_node_kernel<false, false><<<gNode, 256, 0, stream>>>(zb, Wqs, bq, qf,  N_NODES);
    gemm_node_kernel<false, false><<<gNode, 256, 0, stream>>>(zb, Wks, bk, kf,  N_NODES);
    gemm_node_kernel<true,  true ><<<gNode, 256, 0, stream>>>(zb, W1s, b1, tb,  N_NODES);
    gemm_node_kernel<false, false><<<gNode, 256, 0, stream>>>(tb, W2s, b2, gzf, N_NODES);

    // 3) Per-pair sigmoid gate
    attn_kernel<<<E_EDGES / 8, 256, 0, stream>>>(qf, kf, src, dst, af, E_EDGES);

    // 4) Fused h_tilde + predictor GEMM -> edge_logits [E, V] fp32
    dim3 gEdge(E_EDGES / 128, VOCAB / 128);
    edge_logits_kernel<<<gEdge, 256, 0, stream>>>(h_mask, gzf, af, src, dst, Wps, bpred, out);

    // 5) Second tuple output: masked_idx (int32 bits into fp32-typed tail)
    int* outTail = (int*)(out + (size_t)E_EDGES * VOCAB);
    copy_int_kernel<<<(E_EDGES + 255) / 256, 256, 0, stream>>>(masked_idx, outTail, E_EDGES);
}
